// CrossFusion_3779571220897
// MI455X (gfx1250) — matrix operand order repaired and checked by reading, not compiled, not run on hardware
//
#include <hip/hip_runtime.h>
#include <hip/hip_bf16.h>

// ---------------------------------------------------------------------------
// CrossFusion on MI455X (gfx1250, wave32).
// Rank-1 attention (KD=VD=1) => exp-bound streaming softmax; WMMA used for the
// real GEMMs (q/k/v projections with L2-norm + CBN folded into scales).
// B matrices pre-packed into the WMMA per-lane VGPR layout to keep the WMMA
// kernel branch-free; k/v staged to LDS with gfx1250 async loads.
// ---------------------------------------------------------------------------

#define S1N 4096
#define S2N 4096
#define DN  64

// workspace float offsets
#define WS_H     0      // 64
#define WS_MU    64     // 64
#define WS_VAR   128    // 64
#define WS_A     192    // 64   a = (1+dg)*rsqrt(var+eps)
#define WS_B     256    // 64   b = db - a*mu
#define WS_CV    320    // 2    cv[h] = sum_c b[c]*Wv[c][h]
#define WS_KST   324    // 4    kmax0,kmin0,kmax1,kmin1
#define WS_BQ    512    // 512 dwords: packed f16 B for x1 path (2 chunks x 32 lanes x 8)
#define WS_BKV   1024   // 512 dwords: packed f16 B for x2 path
#define WS_Q     1536   // 8192
#define WS_K     9728   // 8192
#define WS_V     17920  // 8192

typedef __attribute__((ext_vector_type(16))) _Float16 v16h;
typedef __attribute__((ext_vector_type(8)))  float    v8f;

#if __has_builtin(__builtin_amdgcn_global_load_async_to_lds_b128)
#define CF_HAS_ASYNC 1
typedef __attribute__((ext_vector_type(4))) int v4i_cf;
typedef __attribute__((address_space(1))) v4i_cf g_v4i;   // global int4
typedef __attribute__((address_space(3))) v4i_cf l_v4i;   // LDS int4
__device__ __forceinline__ void cf_async_cp16(const float* g, float* l) {
  // (global src, lds dst, imm offset, imm cpol)
  __builtin_amdgcn_global_load_async_to_lds_b128((g_v4i*)g, (l_v4i*)l, 0, 0);
}
#endif

// --------------------------------------------------------------------------
// Kernel 1: column stats. block c: h[c]=mean(x1[:,c]); mu,var of x2[:,c].
// --------------------------------------------------------------------------
__global__ __launch_bounds__(256) void cf_colstats(const float* __restrict__ x1,
                                                   const float* __restrict__ x2,
                                                   float* __restrict__ ws) {
  __shared__ float r0[256], r1[256], r2[256];
  const int c = blockIdx.x, tid = threadIdx.x;
  float s1 = 0.f, s2 = 0.f, q2 = 0.f;
  for (int i = tid; i < S1N; i += 256) {
    s1 += x1[i * DN + c];
    float t = x2[i * DN + c];
    s2 += t; q2 += t * t;
  }
  r0[tid] = s1; r1[tid] = s2; r2[tid] = q2;
  __syncthreads();
  for (int off = 128; off > 0; off >>= 1) {
    if (tid < off) {
      r0[tid] += r0[tid + off];
      r1[tid] += r1[tid + off];
      r2[tid] += r2[tid + off];
    }
    __syncthreads();
  }
  if (tid == 0) {
    float h  = r0[0] * (1.f / S1N);
    float mu = r1[0] * (1.f / S2N);
    float vr = r2[0] * (1.f / S2N) - mu * mu;
    ws[WS_H + c] = h; ws[WS_MU + c] = mu; ws[WS_VAR + c] = vr;
  }
}

// --------------------------------------------------------------------------
// Kernel 2: CBN MLP (64x64, tiny). Produces a[c], b[c], cv[h] and the
// pre-packed per-lane f16 B fragments for both WMMA GEMM passes.
// --------------------------------------------------------------------------
__global__ __launch_bounds__(64) void cf_mlp(const float* __restrict__ Wq,
                                             const float* __restrict__ Wk,
                                             const float* __restrict__ Wv,
                                             const float* __restrict__ Wg1,
                                             const float* __restrict__ Wg2,
                                             const float* __restrict__ Wb1,
                                             const float* __restrict__ Wb2,
                                             float* __restrict__ ws) {
  __shared__ float sh[64], t1[64], t2[64], sa[64];
  const int j = threadIdx.x;
  sh[j] = ws[WS_H + j];
  __syncthreads();
  float a1 = 0.f, a2 = 0.f;
  #pragma unroll 8
  for (int i = 0; i < 64; ++i) {
    a1 = fmaf(sh[i], Wg1[i * 64 + j], a1);
    a2 = fmaf(sh[i], Wb1[i * 64 + j], a2);
  }
  t1[j] = fmaxf(a1, 0.f); t2[j] = fmaxf(a2, 0.f);
  __syncthreads();
  float dg = 0.f, db = 0.f;
  #pragma unroll 8
  for (int i = 0; i < 64; ++i) {
    dg = fmaf(t1[i], Wg2[i * 64 + j], dg);
    db = fmaf(t2[i], Wb2[i * 64 + j], db);
  }
  const float var = ws[WS_VAR + j], mu = ws[WS_MU + j];
  const float a = (1.f + dg) * rsqrtf(var + 1e-5f);
  const float b = db - a * mu;
  ws[WS_A + j] = a;
  ws[WS_B + j] = b;
  __syncthreads();                 // t1/t2 no longer needed
  t1[j] = b;                       // stash b for cv reduction
  sa[j] = a;                       // stash a for B packing
  __syncthreads();
  if (j < 2) {
    float cv = 0.f;
    for (int i = 0; i < 64; ++i) cv = fmaf(t1[i], Wv[i * 2 + j], cv);
    ws[WS_CV + j] = cv;
  }
  // ---- pack B fragments in the 16-bit B (32x16) per-lane layout, contiguous k as in A ----
  // dword index within a matrix: c*256 + lane*8 + v ; lane: n=lane&15,
  // half=lane>>4 ; K pair = (kc, kc+1), kc = 2v + 16*half + 32*c.
  unsigned* __restrict__ wsu = (unsigned*)ws;
  for (int idx = j; idx < 1024; idx += 64) {
    const int mat  = idx >> 9;        // 0 = BQ (x1), 1 = BKV (x2)
    const int r    = idx & 511;
    const int c    = r >> 8;
    const int l    = (r >> 3) & 31;
    const int v    = r & 7;
    const int n    = l & 15;
    const int half = l >> 4;
    const int kc   = 2 * v + half * 16 + c * 32;
    float b0 = 0.f, b1 = 0.f;
    if (mat == 0) {
      if (n < 2)      { b0 = Wq[kc * 2 + n]; b1 = Wq[(kc + 1) * 2 + n]; }
    } else {
      if (n < 2)      { b0 = Wk[kc * 2 + n]; b1 = Wk[(kc + 1) * 2 + n]; }
      else if (n < 4) { b0 = sa[kc]     * Wv[kc * 2 + (n - 2)];
                        b1 = sa[kc + 1] * Wv[(kc + 1) * 2 + (n - 2)]; }
    }
    union { _Float16 h[2]; unsigned u; } pk;
    pk.h[0] = (_Float16)b0; pk.h[1] = (_Float16)b1;
    wsu[WS_BQ + idx] = pk.u;
  }
}

// --------------------------------------------------------------------------
// Kernel 3: WMMA projections (branch-free hot path).
// ISX2=false: q = (x1@Wq)*rn1.  ISX2=true: k = (x2@Wk)*rn2,
// v = (x2@(a.*Wv)+cv)*rnv.  One wave per 16-row tile; two 16x16x32 WMMAs.
// --------------------------------------------------------------------------
template <bool ISX2>
__global__ __launch_bounds__(256) void cf_proj(const float* __restrict__ X,
                                               const float* __restrict__ ws,
                                               float* __restrict__ qout,
                                               float* __restrict__ kout,
                                               float* __restrict__ vout) {
  __shared__ float rnA[128];   // 1/||row||
  __shared__ float rnB[128];   // 1/||a.*x2+b|| (x2 path only)
  const int tid = threadIdx.x;
  const int rowbase = blockIdx.x * 128;
  const float* __restrict__ av = ws + WS_A;
  const float* __restrict__ bv = ws + WS_B;

  if (tid < 128) {
    const float* xr = X + (rowbase + tid) * DN;
    float s = 0.f, sv2 = 0.f;
    if (ISX2) {
      #pragma unroll 8
      for (int i = 0; i < DN; ++i) {
        float t = xr[i];
        s = fmaf(t, t, s);
        float u = fmaf(av[i], t, bv[i]);
        sv2 = fmaf(u, u, sv2);
      }
      rnA[tid] = 1.f / fmaxf(sqrtf(s),   1e-12f);
      rnB[tid] = 1.f / fmaxf(sqrtf(sv2), 1e-12f);
    } else {
      #pragma unroll 8
      for (int i = 0; i < DN; ++i) { float t = xr[i]; s = fmaf(t, t, s); }
      rnA[tid] = 1.f / fmaxf(sqrtf(s), 1e-12f);
    }
  }
  __syncthreads();

  const int wave = tid >> 5;
  const int lane = tid & 31;
  const int m    = lane & 15;      // A row
  const int half = lane >> 4;      // lane group
  const int n    = m;              // B/D column for this lane
  const int tilebase = rowbase + wave * 16;

  const v16h* __restrict__ Bp =
      (const v16h*)((const unsigned*)ws + (ISX2 ? WS_BKV : WS_BQ));

  v8f acc = {};
  #pragma unroll
  for (int c = 0; c < 2; ++c) {
    // A layout: lane covers one contiguous 16-float run of the row (k = 16*half + s).
    const float* xr = X + (tilebase + m) * DN + c * 32 + half * 16;
    const float4 p0 = ((const float4*)xr)[0];
    const float4 p1 = ((const float4*)xr)[1];
    const float4 p2 = ((const float4*)(xr + 8))[0];
    const float4 p3 = ((const float4*)(xr + 8))[1];
    v16h A;
    A[0]  = (_Float16)p0.x; A[1]  = (_Float16)p0.y;
    A[2]  = (_Float16)p0.z; A[3]  = (_Float16)p0.w;
    A[4]  = (_Float16)p1.x; A[5]  = (_Float16)p1.y;
    A[6]  = (_Float16)p1.z; A[7]  = (_Float16)p1.w;
    A[8]  = (_Float16)p2.x; A[9]  = (_Float16)p2.y;
    A[10] = (_Float16)p2.z; A[11] = (_Float16)p2.w;
    A[12] = (_Float16)p3.x; A[13] = (_Float16)p3.y;
    A[14] = (_Float16)p3.z; A[15] = (_Float16)p3.w;
    const v16h B = Bp[c * 32 + lane];
    acc = __builtin_amdgcn_wmma_f32_16x16x32_f16(false, A, false, B,
                                                 (short)0, acc, false, false);
  }

  // D layout: vgpr r holds row m = r + 8*half, column n = lane&15.
  const float cvv = (ISX2 && n >= 2 && n < 4) ? ws[WS_CV + (n - 2)] : 0.f;
  #pragma unroll
  for (int r = 0; r < 8; ++r) {
    const int mr   = r + half * 8;
    const int grow = tilebase + mr;
    const int lr   = wave * 16 + mr;
    const float val = acc[r];
    if (ISX2) {
      if (n < 2)      kout[grow * 2 + n]       = val * rnA[lr];
      else if (n < 4) vout[grow * 2 + (n - 2)] = (val + cvv) * rnB[lr];
    } else {
      if (n < 2)      qout[grow * 2 + n]       = val * rnA[lr];
    }
  }
}

// --------------------------------------------------------------------------
// Kernel 4: per-head kmax/kmin (exact softmax max since logits are rank-1).
// --------------------------------------------------------------------------
__global__ __launch_bounds__(256) void cf_kminmax(const float* __restrict__ kv,
                                                  float* __restrict__ kstat) {
  __shared__ float red[1024];
  const int tid = threadIdx.x;
  float mx0 = -3.402823466e38f, mn0 = 3.402823466e38f;
  float mx1 = -3.402823466e38f, mn1 = 3.402823466e38f;
  for (int t = tid; t < S2N; t += 256) {
    float k0 = kv[2 * t], k1 = kv[2 * t + 1];
    mx0 = fmaxf(mx0, k0); mn0 = fminf(mn0, k0);
    mx1 = fmaxf(mx1, k1); mn1 = fminf(mn1, k1);
  }
  red[tid] = mx0; red[256 + tid] = mn0; red[512 + tid] = mx1; red[768 + tid] = mn1;
  __syncthreads();
  for (int off = 128; off > 0; off >>= 1) {
    if (tid < off) {
      red[tid]       = fmaxf(red[tid],       red[tid + off]);
      red[256 + tid] = fminf(red[256 + tid], red[256 + tid + off]);
      red[512 + tid] = fmaxf(red[512 + tid], red[512 + tid + off]);
      red[768 + tid] = fminf(red[768 + tid], red[768 + tid + off]);
    }
    __syncthreads();
  }
  if (tid == 0) {
    kstat[0] = red[0]; kstat[1] = red[256];
    kstat[2] = red[512]; kstat[3] = red[768];
  }
}

// --------------------------------------------------------------------------
// Kernel 5: streaming rank-1 softmax + proj_o + sigmoid. One thread per s.
// k/v staged through LDS in 2048-row chunks via gfx1250 async loads.
// --------------------------------------------------------------------------
__global__ __launch_bounds__(256) void cf_attn(const float* __restrict__ qv,
                                               const float* __restrict__ kv,
                                               const float* __restrict__ vv,
                                               const float* __restrict__ kstat,
                                               const float* __restrict__ Wo,
                                               const float* __restrict__ bo,
                                               float* __restrict__ out) {
  __shared__ float sk[4096];
  __shared__ float sv[4096];
  const int tid = threadIdx.x;
  const int s   = blockIdx.x * 256 + tid;

  const float q0 = qv[2 * s], q1 = qv[2 * s + 1];
  const float m0 = fmaxf(q0 * kstat[0], q0 * kstat[1]);
  const float m1 = fmaxf(q1 * kstat[2], q1 * kstat[3]);
  const float LOG2E = 1.4426950408889634f;
  const float a0 = q0 * LOG2E, c0 = m0 * LOG2E;
  const float a1 = q1 * LOG2E, c1 = m1 * LOG2E;

  float num0 = 0.f, den0 = 0.f, num1 = 0.f, den1 = 0.f;
  for (int chunk = 0; chunk < 2; ++chunk) {
    const int base = chunk * 4096;   // float offset into kv/vv
    __syncthreads();                 // protect previous chunk's reads
#ifdef CF_HAS_ASYNC
    for (int i = tid * 4; i < 4096; i += 256 * 4) {
      cf_async_cp16(kv + base + i, sk + i);
      cf_async_cp16(vv + base + i, sv + i);
    }
    asm volatile("s_wait_asynccnt 0x0" ::: "memory");
#else
    for (int i = tid; i < 4096; i += 256) {
      sk[i] = kv[base + i];
      sv[i] = vv[base + i];
    }
#endif
    __syncthreads();
    if (chunk == 0) {                // warm L2/WGP$ for the next chunk
      __builtin_prefetch(kv + 4096 + tid * 16, 0, 1);
      __builtin_prefetch(vv + 4096 + tid * 16, 0, 1);
    }
    #pragma unroll 4
    for (int t = 0; t < 2048; ++t) {
      float k0 = sk[2 * t], k1 = sk[2 * t + 1];
      float v0 = sv[2 * t], v1 = sv[2 * t + 1];
      float e0 = exp2f(fmaf(a0, k0, -c0));
      float e1 = exp2f(fmaf(a1, k1, -c1));
      num0 = fmaf(e0, v0, num0); den0 += e0;
      num1 = fmaf(e1, v1, num1); den1 += e1;
    }
  }
  const float o0 = num0 / den0;
  const float o1 = num1 / den1;
  const float z0 = fmaf(o0, Wo[0], fmaf(o1, Wo[2], bo[0]));
  const float z1 = fmaf(o0, Wo[1], fmaf(o1, Wo[3], bo[1]));
  out[2 * s]     = 1.f / (1.f + __expf(-z0));
  out[2 * s + 1] = 1.f / (1.f + __expf(-z1));
}

// --------------------------------------------------------------------------
extern "C" void kernel_launch(void* const* d_in, const int* in_sizes, int n_in,
                              void* d_out, int out_size, void* d_ws, size_t ws_size,
                              hipStream_t stream) {
  (void)in_sizes; (void)n_in; (void)out_size; (void)ws_size;
  const float* x1  = (const float*)d_in[0];
  const float* x2  = (const float*)d_in[1];
  const float* Wq  = (const float*)d_in[2];
  const float* Wk  = (const float*)d_in[3];
  const float* Wv  = (const float*)d_in[4];
  const float* Wo  = (const float*)d_in[5];
  const float* bo  = (const float*)d_in[6];
  const float* Wg1 = (const float*)d_in[7];
  const float* Wg2 = (const float*)d_in[8];
  const float* Wb1 = (const float*)d_in[9];
  const float* Wb2 = (const float*)d_in[10];
  float* ws  = (float*)d_ws;
  float* out = (float*)d_out;

  cf_colstats<<<64, 256, 0, stream>>>(x1, x2, ws);
  cf_mlp<<<1, 64, 0, stream>>>(Wq, Wk, Wv, Wg1, Wg2, Wb1, Wb2, ws);
  cf_proj<false><<<32, 256, 0, stream>>>(x1, ws, ws + WS_Q, ws + WS_K, ws + WS_V);
  cf_proj<true><<<32, 256, 0, stream>>>(x2, ws, ws + WS_Q, ws + WS_K, ws + WS_V);
  cf_kminmax<<<1, 256, 0, stream>>>(ws + WS_K, ws + WS_KST);
  cf_attn<<<16, 256, 0, stream>>>(ws + WS_Q, ws + WS_K, ws + WS_V,
                                  ws + WS_KST, Wo, bo, out);
}
